// PLE_Module_21311627722882
// MI455X (gfx1250) — compile-verified, matrix operand order checked by reading, not run on hardware
//
#include <hip/hip_runtime.h>

#define EPS 1e-5f

typedef __attribute__((ext_vector_type(16))) __bf16 bf16x16;
typedef __attribute__((ext_vector_type(8)))  float  f32x8;

union Frag {
  bf16x16 v;
  unsigned short u[16];
  uint4 q[2];
};

__device__ __forceinline__ unsigned short f2bf(float f) {
  unsigned u = __float_as_uint(f);
  u += 0x7FFFu + ((u >> 16) & 1u);   // round-to-nearest-even
  return (unsigned short)(u >> 16);
}

// element(e)/lane -> K mapping of 16-bit WMMA A/B fragments (ISA 7.12.2):
// lane<16: e0..7 -> K0..7,  e8..15 -> K16..23 ; lane>=16: +8
__device__ __forceinline__ int kmap(int e, int lane) {
  return (e & 7) + ((e >> 3) << 4) + ((lane >> 4) << 3);
}

__device__ __forceinline__ f32x8 wmma_bf16(const Frag& A, const Frag& B, f32x8 C) {
  return __builtin_amdgcn_wmma_f32_16x16x32_bf16(false, A.v, false, B.v,
                                                 (short)0, C, false, false);
}

// ---------------------------------------------------------------------------
// Kernel 0: fold BN into weights, convert to bf16 fragment layouts.
// Segments (element counts):
//   [0,16384)          WcatFrag  [mt4][kc8][lane32][e16]   (theta rows 0-31, phi 32-63)
//   [16384,16448)      biasCat[64]
//   [16448,81984)      Wg1B      [kc8][ot16][lane32][e16]  B-frag of Wg[:, :256]^T
//   [81984,147520)     Wg2A      [ot16][kc8][lane32][e16]  A-frag of Wg[:, 256:]
//   [147520,147776)    gbias[256]
//   [147776,221504)    Ws1A      [k3][ut6][kc8][lane32][e16]
//   [221504,221600)    s1bias[96]
//   [221600,221697)    ws2eff[96] + s2bias at [96]
// ---------------------------------------------------------------------------
__global__ void prep_weights(
    const float* Wt, const float* tg, const float* tb, const float* tm, const float* tv,
    const float* Wp, const float* pg, const float* pb, const float* pm, const float* pv,
    const float* Wg, const float* gg, const float* gb, const float* gm, const float* gv,
    const float* Ws1, const float* s1g, const float* s1b, const float* s1m, const float* s1v,
    const float* Ws2, const float* s2g, const float* s2b, const float* s2m, const float* s2v,
    unsigned short* wcat, float* biascat,
    unsigned short* wg1B, unsigned short* wg2A, float* gbias,
    unsigned short* ws1A, float* s1bias, float* ws2eff)
{
  int i = blockIdx.x * 256 + threadIdx.x;
  if (i < 16384) {
    int e = i & 15, lane = (i >> 4) & 31, kc = (i >> 9) & 7, mt = i >> 12;
    int o = mt * 16 + (lane & 15);
    int c = kc * 32 + kmap(e, lane);
    float v;
    if (o < 32) v = tg[o] * rsqrtf(tv[o] + EPS) * Wt[o * 256 + c];
    else { int oo = o - 32; v = pg[oo] * rsqrtf(pv[oo] + EPS) * Wp[oo * 256 + c]; }
    wcat[i] = f2bf(v);
  } else if (i < 16448) {
    int o = i - 16384;
    if (o < 32) biascat[o] = tb[o] - tm[o] * (tg[o] * rsqrtf(tv[o] + EPS));
    else { int oo = o - 32; biascat[o] = pb[oo] - pm[oo] * (pg[oo] * rsqrtf(pv[oo] + EPS)); }
  } else if (i < 81984) {
    int j = i - 16448;
    int e = j & 15, lane = (j >> 4) & 31, ot = (j >> 9) & 15, kc = j >> 13;
    int o = ot * 16 + (lane & 15);
    int n = kc * 32 + kmap(e, lane);
    wg1B[j] = f2bf(gg[o] * rsqrtf(gv[o] + EPS) * Wg[o * 512 + n]);
  } else if (i < 147520) {
    int j = i - 81984;
    int e = j & 15, lane = (j >> 4) & 31, kc = (j >> 9) & 7, ot = j >> 12;
    int o = ot * 16 + (lane & 15);
    int m = kc * 32 + kmap(e, lane);
    wg2A[j] = f2bf(gg[o] * rsqrtf(gv[o] + EPS) * Wg[o * 512 + 256 + m]);
  } else if (i < 147776) {
    int o = i - 147520;
    gbias[o] = gb[o] - gm[o] * (gg[o] * rsqrtf(gv[o] + EPS));
  } else if (i < 221504) {
    int j = i - 147776;
    int e = j & 15, lane = (j >> 4) & 31, kc = (j >> 9) & 7;
    int rest = j >> 12, ut = rest % 6, k = rest / 6;
    int u = ut * 16 + (lane & 15);
    int o = kc * 32 + kmap(e, lane);
    ws1A[j] = f2bf(s1g[u] * rsqrtf(s1v[u] + EPS) * Ws1[u * 768 + k * 256 + o]);
  } else if (i < 221600) {
    int u = i - 221504;
    s1bias[u] = s1b[u] - s1m[u] * (s1g[u] * rsqrtf(s1v[u] + EPS));
  } else if (i < 221697) {
    int u = i - 221600;
    float sc = s2g[0] * rsqrtf(s2v[0] + EPS);
    if (u < 96) ws2eff[u] = sc * Ws2[u];
    else        ws2eff[96] = s2b[0] - s2m[0] * sc;
  }
}

// ---------------------------------------------------------------------------
// Kernel 1: theta/phi = relu(bn(Wcat @ x_quadrant)), one block per (q,b,t).
// 8 waves, 64x256x256 GEMM via bf16 WMMA; exports theta as A-frags and phi as
// B-frags (both share the same storage transform) to global scratch.
// LDS: xB frags 16KB + plain f32 result 64KB (dynamic, 80KB).
// ---------------------------------------------------------------------------
__global__ void __launch_bounds__(256, 1)
theta_phi_kernel(const float* __restrict__ x,
                 const unsigned short* __restrict__ wcat,
                 const float* __restrict__ biascat,
                 unsigned short* __restrict__ thetaF,
                 unsigned short* __restrict__ phiF)
{
  extern __shared__ char smem[];
  unsigned short* xB = (unsigned short*)smem;          // [st16][lane32][e16]
  float* plain = (float*)(smem + 16384);               // [64][256]

  int blk = blockIdx.x;
  int t = blk & 7, b = (blk >> 3) & 15, q = blk >> 7;
  int h0 = (q >> 1) << 4, w0 = (q & 1) << 4;
  int tid = threadIdx.x, lane = tid & 31, wave = tid >> 5;
  int laneHi = lane >> 4, ln = lane & 15;

  f32x8 acc[8] = {};

  for (int kc = 0; kc < 8; ++kc) {
    // stage x chunk (32 channels x 256 positions) into B-fragment layout
    #pragma unroll
    for (int rr = 0; rr < 2; ++rr) {
      int rid = tid * 2 + rr;
      int cl = rid >> 4;          // local channel 0..31 (K index)
      int hp = rid & 15;          // quadrant row == s-tile index
      int c = kc * 32 + cl;
      const float* src = x + ((((size_t)b * 256 + c) * 8 + t) * 1024) + (h0 + hp) * 32 + w0;
      int lh = (cl >> 3) & 1;
      int e = (cl & 7) | ((cl >> 4) << 3);
      #pragma unroll
      for (int wp = 0; wp < 16; ++wp)
        xB[((hp * 32) + lh * 16 + wp) * 16 + e] = f2bf(src[wp]);
    }
    __syncthreads();
    #pragma unroll
    for (int j = 0; j < 8; ++j) {
      int tile = wave * 8 + j;
      int mt = tile >> 4, nt = tile & 15;
      Frag fa, fb;
      const uint4* ap = (const uint4*)(wcat + ((mt * 8 + kc) * 32 + lane) * 16);
      fa.q[0] = ap[0]; fa.q[1] = ap[1];
      const uint4* bp = (const uint4*)(xB + ((nt * 32) + lane) * 16);
      fb.q[0] = bp[0]; fb.q[1] = bp[1];
      acc[j] = wmma_bf16(fa, fb, acc[j]);
    }
    __syncthreads();
  }

  // bias + relu -> plain f32 [o][s]
  #pragma unroll
  for (int j = 0; j < 8; ++j) {
    int tile = wave * 8 + j, mt = tile >> 4, nt = tile & 15;
    #pragma unroll
    for (int r = 0; r < 8; ++r) {
      int o = mt * 16 + r + 8 * laneHi;
      int s = nt * 16 + ln;
      plain[o * 256 + s] = fmaxf(acc[j][r] + biascat[o], 0.0f);
    }
  }
  __syncthreads();

  // export to fragment layout in global scratch (theta A-frag / phi B-frag,
  // identical index transform: frag[st][lp][e] = M[kmap(e,lp)][st*16+lp%16])
  size_t fbase = ((((size_t)q * 16 + b) * 8 + t) * 8192);
  #pragma unroll
  for (int ii = 0; ii < 4; ++ii) {
    int it = tid + 256 * ii;
    int M = it >> 9, st = (it >> 5) & 15, lp = it & 31;
    int s = st * 16 + (lp & 15);
    unsigned short* dst = (M ? phiF : thetaF) + fbase + (st * 32 + lp) * 16;
    #pragma unroll
    for (int e = 0; e < 16; ++e) {
      int ic = kmap(e, lp);
      dst[e] = f2bf(plain[(M * 32 + ic) * 256 + s]);
    }
  }
}

// ---------------------------------------------------------------------------
// Kernel 2: fully fused PLE + w_spatial + gating. One block per (i,b,t),
// 16 waves. For each of 3 partners: Gs in LDS (bf16), PLE tiles computed and
// consumed in-place into persistent hmid WMMA accumulators, then gate+scale.
// LDS: thA 16K | phB 16K | Gs 128K | PLEfrag 128K | scratch 16K | gate 1K = 305KB
// ---------------------------------------------------------------------------
__global__ void __launch_bounds__(512, 1)
ple_fused_kernel(const float* __restrict__ x, float* __restrict__ out,
                 const unsigned short* __restrict__ thetaF,
                 const unsigned short* __restrict__ phiF,
                 const unsigned short* __restrict__ wg1B,
                 const unsigned short* __restrict__ wg2A,
                 const unsigned short* __restrict__ ws1A,
                 const float* __restrict__ gbias,
                 const float* __restrict__ s1bias,
                 const float* __restrict__ ws2eff)
{
  extern __shared__ char smem[];
  unsigned short* thA = (unsigned short*)smem;             // theta A-frags
  unsigned short* phB = (unsigned short*)(smem + 16384);   // phi B-frags
  unsigned short* Gs  = (unsigned short*)(smem + 32768);   // plain [256][256] bf16
  unsigned short* PLE = (unsigned short*)(smem + 163840);  // B-frags [kc8][st16][lane32][e16]
  float* scratch = (float*)(smem + 294912);                // per-wave 16x16 transpose tile
  float* gate = (float*)(smem + 311296);                   // [256]

  int blk = blockIdx.x;
  int t = blk & 7, b = (blk >> 3) & 15, iq = blk >> 7;
  int tid = threadIdx.x, lane = tid & 31, wave = tid >> 5;
  int laneHi = lane >> 4, ln = lane & 15;
  int st = wave;                  // this wave owns s-tile == wave id
  float* scr = scratch + wave * 256;

  // warm L2 for the shared weight fragments (global_prefetch_b8)
  for (int p = tid; p < 2048; p += 512) {
    __builtin_prefetch(wg1B + p * 64, 0, 1);
    __builtin_prefetch(wg2A + p * 64, 0, 1);
  }

  // load theta fragments for quadrant iq (16KB, already in frag layout)
  {
    const uint4* src = (const uint4*)(thetaF + ((((size_t)iq * 16 + b) * 8 + t) * 8192));
    uint4* dst = (uint4*)thA;
    dst[tid * 2] = src[tid * 2];
    dst[tid * 2 + 1] = src[tid * 2 + 1];
  }

  f32x8 acc3[6] = {};   // persistent hmid accumulators: u in [ut*16,ut*16+16), s-tile = wave

  for (int k = 0; k < 3; ++k) {
    int jq = (k >= iq) ? k + 1 : k;          // partners in ascending order, skipping iq
    __syncthreads();                          // protect Gs/phB from prior readers
    {
      const uint4* src = (const uint4*)(phiF + ((((size_t)jq * 16 + b) * 8 + t) * 8192));
      uint4* dst = (uint4*)phB;
      dst[tid * 2] = src[tid * 2];
      dst[tid * 2 + 1] = src[tid * 2 + 1];
    }
    __syncthreads();

    // phase 1: Gs[m,n] = theta @ phi  (K=32 -> one WMMA per 16x16 tile)
    Frag fa;
    { const uint4* ap = (const uint4*)(thA + (wave * 32 + lane) * 16);
      fa.q[0] = ap[0]; fa.q[1] = ap[1]; }
    #pragma unroll
    for (int nt = 0; nt < 16; ++nt) {
      Frag fb;
      const uint4* bp = (const uint4*)(phB + (nt * 32 + lane) * 16);
      fb.q[0] = bp[0]; fb.q[1] = bp[1];
      f32x8 d = {};
      d = wmma_bf16(fa, fb, d);
      #pragma unroll
      for (int r = 0; r < 8; ++r) {
        int m = wave * 16 + r + 8 * laneHi;
        int n = nt * 16 + ln;
        Gs[m * 256 + n] = f2bf(d[r]);
      }
    }
    __syncthreads();

    // phase 2: PLE[o, s-tile] = relu(Gs[s,:]·Wg1^T + Wg2·Gs[:,s] + gbias)
    #pragma unroll 1
    for (int ot = 0; ot < 16; ++ot) {
      f32x8 acc1 = {}, acc2 = {};
      #pragma unroll
      for (int kc = 0; kc < 8; ++kc) {         // A[s,o] part
        Frag ag, bw;
        int srow = st * 16 + ln;
        #pragma unroll
        for (int e = 0; e < 16; ++e)
          ag.u[e] = Gs[srow * 256 + kc * 32 + kmap(e, lane)];
        const uint4* bp = (const uint4*)(wg1B + ((kc * 16 + ot) * 32 + lane) * 16);
        bw.q[0] = bp[0]; bw.q[1] = bp[1];
        acc1 = wmma_bf16(ag, bw, acc1);
      }
      #pragma unroll
      for (int kc = 0; kc < 8; ++kc) {         // B'[o,s] part
        Frag aw, bg;
        const uint4* ap = (const uint4*)(wg2A + ((ot * 8 + kc) * 32 + lane) * 16);
        aw.q[0] = ap[0]; aw.q[1] = ap[1];
        int scol = st * 16 + ln;
        #pragma unroll
        for (int e = 0; e < 16; ++e)
          bg.u[e] = Gs[(kc * 32 + kmap(e, lane)) * 256 + scol];
        acc2 = wmma_bf16(aw, bg, acc2);
      }
      // in-wave transpose of acc1 via private LDS tile (DS is in-order per wave)
      #pragma unroll
      for (int r = 0; r < 8; ++r)
        scr[ln * 16 + (r + 8 * laneHi)] = acc1[r];   // scr[o][s] = A[s,o]
      asm volatile("s_wait_dscnt 0x0" ::: "memory");
      #pragma unroll
      for (int r = 0; r < 8; ++r) {
        int oLoc = r + 8 * laneHi;
        int o = ot * 16 + oLoc;
        float v = fmaxf(acc2[r] + scr[oLoc * 16 + ln] + gbias[o], 0.0f);
        int kc2 = o >> 5, ko = o & 31;
        int e2 = (ko & 7) | (((ko >> 4) & 1) << 3);
        int lh2 = (ko >> 3) & 1;
        PLE[((kc2 * 16 + st) * 32 + lh2 * 16 + ln) * 16 + e2] = f2bf(v);
      }
    }
    asm volatile("s_wait_dscnt 0x0" ::: "memory");

    // phase 3: hmid[u, s-tile] += Ws1[:, k-block] @ PLE  (wave-private column)
    #pragma unroll
    for (int ut = 0; ut < 6; ++ut) {
      #pragma unroll
      for (int kc = 0; kc < 8; ++kc) {
        Frag aw, fp;
        const uint4* ap = (const uint4*)(ws1A + (((k * 6 + ut) * 8 + kc) * 32 + lane) * 16);
        aw.q[0] = ap[0]; aw.q[1] = ap[1];
        const uint4* bp = (const uint4*)(PLE + ((kc * 16 + st) * 32 + lane) * 16);
        fp.q[0] = bp[0]; fp.q[1] = bp[1];
        acc3[ut] = wmma_bf16(aw, fp, acc3[ut]);
      }
    }
  }

  // finalize: h = relu(hmid + s1bias); wsum = Ws2eff·h + s2bias; gate = sigmoid
  float sum = 0.0f;
  #pragma unroll
  for (int ut = 0; ut < 6; ++ut) {
    #pragma unroll
    for (int r = 0; r < 8; ++r) {
      int u = ut * 16 + r + 8 * laneHi;
      float h = fmaxf(acc3[ut][r] + s1bias[u], 0.0f);
      sum += ws2eff[u] * h;
    }
  }
  float other = __shfl_xor(sum, 16, 32);       // combine the two row-halves (wave32)
  float wsum = sum + other + ws2eff[96];
  float g = 1.0f / (1.0f + expf(-wsum));
  if (lane < 16) gate[wave * 16 + lane] = g;
  __syncthreads();

  // gated copy of the x quadrant to the output (quadrants map back in place)
  int h0 = (iq >> 1) << 4, w0 = (iq & 1) << 4;
  for (int idx = tid; idx < 65536; idx += 512) {
    int s = idx & 255, c = idx >> 8;
    int hp = s >> 4, wp = s & 15;
    size_t off = ((((size_t)b * 256 + c) * 8 + t) * 1024) + (h0 + hp) * 32 + (w0 + wp);
    out[off] = gate[s] * x[off];
  }
}

// ---------------------------------------------------------------------------
extern "C" void kernel_launch(void* const* d_in, const int* in_sizes, int n_in,
                              void* d_out, int out_size, void* d_ws, size_t ws_size,
                              hipStream_t stream) {
  const float* x   = (const float*)d_in[0];
  const float* Wt  = (const float*)d_in[1];
  const float* tg  = (const float*)d_in[2];
  const float* tb  = (const float*)d_in[3];
  const float* tm  = (const float*)d_in[4];
  const float* tv  = (const float*)d_in[5];
  const float* Wp  = (const float*)d_in[6];
  const float* pg  = (const float*)d_in[7];
  const float* pb  = (const float*)d_in[8];
  const float* pm  = (const float*)d_in[9];
  const float* pv  = (const float*)d_in[10];
  const float* Wg  = (const float*)d_in[11];
  const float* gg  = (const float*)d_in[12];
  const float* gb  = (const float*)d_in[13];
  const float* gm  = (const float*)d_in[14];
  const float* gv  = (const float*)d_in[15];
  const float* Ws1 = (const float*)d_in[16];
  const float* s1g = (const float*)d_in[17];
  const float* s1b = (const float*)d_in[18];
  const float* s1m = (const float*)d_in[19];
  const float* s1v = (const float*)d_in[20];
  const float* Ws2 = (const float*)d_in[21];
  const float* s2g = (const float*)d_in[22];
  const float* s2b = (const float*)d_in[23];
  const float* s2m = (const float*)d_in[24];
  const float* s2v = (const float*)d_in[25];

  char* ws = (char*)d_ws;
  unsigned short* wcat   = (unsigned short*)(ws + 0);        // 32768 B
  float*          biascat= (float*)(ws + 32768);             // 256 B
  unsigned short* wg1B   = (unsigned short*)(ws + 33024);    // 131072 B
  unsigned short* wg2A   = (unsigned short*)(ws + 164096);   // 131072 B
  float*          gbias  = (float*)(ws + 295168);            // 1024 B
  unsigned short* ws1A   = (unsigned short*)(ws + 296192);   // 147456 B
  float*          s1bias = (float*)(ws + 443648);            // 384 B
  float*          ws2eff = (float*)(ws + 444032);            // 512 B
  unsigned short* thetaF = (unsigned short*)(ws + (1u << 20));            // 8 MiB
  unsigned short* phiF   = (unsigned short*)(ws + (1u << 20) + (8u << 20)); // 8 MiB

  prep_weights<<<867, 256, 0, stream>>>(
      Wt, tg, tb, tm, tv, Wp, pg, pb, pm, pv, Wg, gg, gb, gm, gv,
      Ws1, s1g, s1b, s1m, s1v, Ws2, s2g, s2b, s2m, s2v,
      wcat, biascat, wg1B, wg2A, gbias, ws1A, s1bias, ws2eff);

  theta_phi_kernel<<<512, 256, 81920, stream>>>(x, wcat, biascat, thetaF, phiF);

  ple_fused_kernel<<<512, 512, 312320, stream>>>(
      x, (float*)d_out, thetaF, phiF, wg1B, wg2A, ws1A, gbias, s1bias, ws2eff);

  (void)in_sizes; (void)n_in; (void)out_size; (void)ws_size;
}
